// ConvLoRA_137438953649
// MI455X (gfx1250) — matrix operand order repaired and checked by reading, not compiled, not run on hardware
//
#include <hip/hip_runtime.h>

typedef __attribute__((ext_vector_type(16))) __bf16 v16bf;
typedef __attribute__((ext_vector_type(8)))  float  v8f;

#define NB   16
#define CIN  256
#define COUT 256
#define HW   128

// ---------------------------------------------------------------------------
// Kernel A: embed MLPs -> coff1/coff2, fuse LoRA into per-sample conv weights.
// Wp is written PRE-SWIZZLED in WMMA A-fragment order:
//   Wp[b][tap(9)][mtile(16)][kchunk(8)][lane(32)][elem(16)]  (bf16)
// so each lane's A fragment in the conv kernel is ONE contiguous 32-byte load.
// Fragment mapping (contiguous k order, same as the B reads; 16x32 MxK):
//   lane = half*16 + mrow,  cout = mtile*16 + mrow
//   elem e -> klocal = half*16 + e,  cin = kchunk*32+klocal
// ---------------------------------------------------------------------------
__global__ __launch_bounds__(256)
void prep_weights(const float* __restrict__ wms,  const float* __restrict__ conv_w,
                  const float* __restrict__ e1_w1, const float* __restrict__ e1_b1,
                  const float* __restrict__ e1_w2, const float* __restrict__ e1_b2,
                  const float* __restrict__ e2_w1, const float* __restrict__ e2_b1,
                  const float* __restrict__ e2_w2, const float* __restrict__ e2_b2,
                  const float* __restrict__ lA1,  const float* __restrict__ lA2,
                  const float* __restrict__ lB,   __bf16* __restrict__ Wp)
{
    __shared__ float sh_h1[128], sh_h2[128];
    __shared__ float sh_c1[256], sh_c2[256];
    __shared__ float sh_T[16 * 768];          // T = coff1@A1 + coff2@A2 (lora_B shared)

    const int b = blockIdx.x;
    const int t = threadIdx.x;
    const float* wm1 = wms + b * 32;
    const float* wm2 = wms + NB * 32 + b * 32;

    // Linear(32,128) + LeakyReLU(0.2): threads 0-127 branch1, 128-255 branch2
    if (t < 128) {
        float a = e1_b1[t];
        for (int i = 0; i < 32; ++i) a += wm1[i] * e1_w1[t * 32 + i];
        sh_h1[t] = (a >= 0.f) ? a : 0.2f * a;
    } else {
        const int j = t - 128;
        float a = e2_b1[j];
        for (int i = 0; i < 32; ++i) a += wm2[i] * e2_w1[j * 32 + i];
        sh_h2[j] = (a >= 0.f) ? a : 0.2f * a;
    }
    __syncthreads();

    // Linear(128,256) -> coff (flattened 16x16)
    {
        float a1 = e1_b2[t], a2 = e2_b2[t];
        for (int j = 0; j < 128; ++j) {
            a1 += sh_h1[j] * e1_w2[t * 128 + j];
            a2 += sh_h2[j] * e2_w2[t * 128 + j];
        }
        sh_c1[t] = a1; sh_c2[t] = a2;
    }
    __syncthreads();

    // T[r,c] = sum_q coff1[r,q]*A1[q,c] + coff2[r,q]*A2[q,c]
    for (int idx = t; idx < 16 * 768; idx += 256) {
        const int r = idx / 768, c = idx % 768;
        float s = 0.f;
        for (int q = 0; q < 16; ++q)
            s += sh_c1[r * 16 + q] * lA1[q * 768 + c]
               + sh_c2[r * 16 + q] * lA2[q * 768 + c];
        sh_T[idx] = s;
    }
    __syncthreads();

    // W_b[flat] = conv_w[flat] + sum_r lora_B[p,r]*T[r,c],
    // p = flat/768, c = flat%768  (exact replica of the reference reshape)
    for (int f = t; f < COUT * CIN * 9; f += 256) {
        const int cout = f / 2304, rem = f % 2304;
        const int cin = rem / 9, t9 = rem % 9;           // t9 = kh*3+kw = tap
        const int p = f / 768, c = f % 768;
        float d = conv_w[f];
        for (int r = 0; r < 16; ++r) d += lB[p * 16 + r] * sh_T[r * 768 + c];

        // scatter into fragment-swizzled layout
        const int mt = cout >> 4, mrow = cout & 15;
        const int cc = cin >> 5,  kl   = cin & 31;
        int half, e;
        if (kl < 16) { half = 0; e = kl; }
        else         { half = 1; e = kl - 16; }
        const int lane = half * 16 + mrow;
        Wp[((((b * 9 + t9) * 16 + mt) * 8 + cc) * 32 + lane) * 16 + e] = (__bf16)d;
    }
}

// ---------------------------------------------------------------------------
// Kernel B: implicit-GEMM 3x3 conv, bf16 WMMA, f32 accumulate.
// Block = 256 threads (8 waves, 4x2 wave grid), tile = 128 couts x 128 pixels
// (one full y row -> halo only in x and rows y-1..y+1).
// ---------------------------------------------------------------------------
__global__ __launch_bounds__(256)
void conv_wmma(const float* __restrict__ x, const float* __restrict__ bias,
               const __bf16* __restrict__ Wp, float* __restrict__ out)
{
    __shared__ __align__(32) __bf16 xs[3 * 132 * 32];  // [row 0..2][col 0..131][cin32]

    const int y        = blockIdx.x;
    const int coutBase = blockIdx.y * 128;
    const int b        = blockIdx.z;

    const int lane = threadIdx.x & 31;
    const int wave = threadIdx.x >> 5;
    const int wm   = wave & 3;            // 4 waves across M (couts)
    const int wn   = wave >> 2;           // 2 waves across N (pixels)
    const int l16  = lane & 15;
    const int half = lane >> 4;

    v8f acc[2][4];
    for (int i = 0; i < 2; ++i)
        for (int j = 0; j < 4; ++j)
            acc[i][j] = (v8f){};

    for (int cc = 0; cc < 8; ++cc) {               // cin chunks of 32
        const int cinBase = cc * 32;

        __syncthreads();                           // protect previous chunk reads
        // Stage x rows y-1..y+1, cols -1..128, 32 cins; f32 -> bf16 transpose.
        for (int idx = (int)threadIdx.x; idx < 3 * 130 * 32; idx += 256) {
            const int ci  = idx / 390;
            const int rem = idx % 390;
            const int r3  = rem / 130;
            const int col = rem % 130;
            const int gy  = y + r3 - 1;
            const int gx  = col - 1;
            float v = 0.f;
            if ((unsigned)gy < 128u && (unsigned)gx < 128u)
                v = x[(((b * CIN) + cinBase + ci) * HW + gy) * HW + gx];
            xs[(r3 * 132 + col) * 32 + ci] = (__bf16)v;
        }
        __syncthreads();

        // Prefetch next chunk's x lines into cache while WMMAs run
        // (global_prefetch_b8; ~5 cachelines per (ci,row) of 130 floats).
        if (cc < 7) {
            const int nb2 = cinBase + 32;
            for (int p = (int)threadIdx.x; p < 3 * 32 * 5; p += 256) {
                const int ci  = p / 15;
                const int rem = p % 15;
                const int r3  = rem / 5;
                const int seg = rem % 5;
                const int gy  = y + r3 - 1;
                if ((unsigned)gy < 128u) {
                    const int gx = seg * 32 > 127 ? 127 : seg * 32;
                    __builtin_prefetch(&x[(((b * CIN) + nb2 + ci) * HW + gy) * HW + gx], 0, 1);
                }
            }
        }

#pragma unroll
        for (int tap = 0; tap < 9; ++tap) {        // (kh,kw) taps
            const int kh = tap / 3, kw = tap % 3;

            // A fragments: single contiguous 32B load per lane (pre-swizzled Wp)
            v16bf afrag[2];
#pragma unroll
            for (int i = 0; i < 2; ++i) {
                const int mt = (coutBase >> 4) + wm * 2 + i;
                afrag[i] = *(const v16bf*)(
                    Wp + ((((b * 9 + tap) * 16 + mt) * 8 + cc) * 32 + lane) * 16);
            }

            // B fragments: contiguous 32B LDS reads (N = lane%16, K = half*16..+15)
            v16bf bfrag[4];
#pragma unroll
            for (int j = 0; j < 4; ++j) {
                const int col = wn * 64 + j * 16 + l16 + kw;   // +kw shifts the window
                bfrag[j] = *(const v16bf*)(&xs[(kh * 132 + col) * 32 + half * 16]);
            }

#pragma unroll
            for (int i = 0; i < 2; ++i)
#pragma unroll
                for (int j = 0; j < 4; ++j)
                    acc[i][j] = __builtin_amdgcn_wmma_f32_16x16x32_bf16(
                        false, afrag[i], false, bfrag[j],
                        (short)0, acc[i][j], false, false);
        }
    }

    // Epilogue: C/D layout -> cout = m0 + half*8 + v, col = n0 + lane%16
    for (int i = 0; i < 2; ++i) {
        for (int j = 0; j < 4; ++j) {
            const int col = wn * 64 + j * 16 + l16;
            for (int v = 0; v < 8; ++v) {
                const int cout = coutBase + wm * 32 + i * 16 + half * 8 + v;
                out[(((b * COUT) + cout) * HW + y) * HW + col] =
                    acc[i][j][v] + bias[cout];
            }
        }
    }
}

// ---------------------------------------------------------------------------
extern "C" void kernel_launch(void* const* d_in, const int* in_sizes, int n_in,
                              void* d_out, int out_size, void* d_ws, size_t ws_size,
                              hipStream_t stream)
{
    const float* x      = (const float*)d_in[0];
    const float* wms    = (const float*)d_in[1];
    const float* conv_w = (const float*)d_in[2];
    const float* conv_b = (const float*)d_in[3];
    const float* e1_w1  = (const float*)d_in[4];
    const float* e1_b1  = (const float*)d_in[5];
    const float* e1_w2  = (const float*)d_in[6];
    const float* e1_b2  = (const float*)d_in[7];
    const float* e2_w1  = (const float*)d_in[8];
    const float* e2_b1  = (const float*)d_in[9];
    const float* e2_w2  = (const float*)d_in[10];
    const float* e2_b2  = (const float*)d_in[11];
    const float* lA1    = (const float*)d_in[12];
    const float* lA2    = (const float*)d_in[13];
    const float* lB     = (const float*)d_in[14];

    __bf16* Wp = (__bf16*)d_ws;   // 16*9*256*256 bf16 = ~18.9 MB

    prep_weights<<<NB, 256, 0, stream>>>(wms, conv_w,
                                         e1_w1, e1_b1, e1_w2, e1_b2,
                                         e2_w1, e2_b1, e2_w2, e2_b2,
                                         lA1, lA2, lB, Wp);

    conv_wmma<<<dim3(HW, COUT / 128, NB), 256, 0, stream>>>(
        x, conv_b, Wp, (float*)d_out);
}
